// GriffinLinear_54623394071033
// MI455X (gfx1250) — matrix operand order repaired and checked by reading, not compiled, not run on hardware
//
#include <hip/hip_runtime.h>
#include <hip/hip_bf16.h>

typedef __attribute__((ext_vector_type(16))) __bf16 v16bf;
typedef __attribute__((ext_vector_type(8)))  __bf16 v8bf;
typedef __attribute__((ext_vector_type(8)))  float  v8f;

#define K_IN      4096
#define N_OUT     11008
#define M_PREFILL 2048
#define M_DECODE  64
#define M_TOTAL   (M_PREFILL + M_DECODE)
#define TOPK      2048

#define BM 128
#define BN 128
#define BK 32
#define LDS_K 40   // padded row stride (bf16 elems): 80B -> conflict-free b128 frag reads

union BU { unsigned short u; __bf16 b; };
union Frag16 { v16bf v; v8bf h[2]; };

// split fp32 -> bf16 hi + bf16 lo (truncation split; a ~= hi + lo)
__device__ inline void split_bf16(float x, __bf16* hi, __bf16* lo) {
    unsigned u  = __float_as_uint(x);
    BU h; h.u = (unsigned short)(u >> 16);
    float fhi = __uint_as_float(u & 0xFFFF0000u);
    float r   = x - fhi;
    BU l; l.u = (unsigned short)(__float_as_uint(r) >> 16);
    *hi = h.b; *lo = l.b;
}

// A fragment (16x32 bf16): lane = half*16 + m ; K chunks {8*half..}, {16+8*half..}
__device__ inline v16bf load_frag_a(const __bf16* base, int lane) {
    const int m = lane & 15, half = lane >> 4;
    const __bf16* row = base + m * LDS_K;
    Frag16 f;
    f.h[0] = *(const v8bf*)(row + 8 * half);
    f.h[1] = *(const v8bf*)(row + 16 + 8 * half);
    return f.v;
}

// B fragment (32x16 bf16), Bs stored as [n][k] (== W layout): lane = half*16 + n ; K chunks {8*half..}, {16+8*half..}
__device__ inline v16bf load_frag_b(const __bf16* base, int lane) {
    const int n = lane & 15, half = lane >> 4;
    const __bf16* row = base + n * LDS_K;
    Frag16 f;
    f.h[0] = *(const v8bf*)(row + 8 * half);
    f.h[1] = *(const v8bf*)(row + 16 + 8 * half);
    return f.v;
}

// ---------- Stage 1: per-row 1/max(||row||,eps)^2 ----------
__global__ __launch_bounds__(256) void row_w_kernel(const float* __restrict__ xp,
                                                    float* __restrict__ wrow) {
    __shared__ float red[256];
    const int r = blockIdx.x;
    const float* row = xp + (size_t)r * K_IN;
    float s = 0.f;
    for (int i = threadIdx.x; i < K_IN; i += 256) { float v = row[i]; s += v * v; }
    red[threadIdx.x] = s;
    __syncthreads();
    for (int st = 128; st > 0; st >>= 1) {
        if (threadIdx.x < st) red[threadIdx.x] += red[threadIdx.x + st];
        __syncthreads();
    }
    if (threadIdx.x == 0) {
        float den = fmaxf(sqrtf(red[0]), 1e-12f);
        wrow[r] = 1.0f / (den * den);
    }
}

// ---------- Stage 2: col_norm[d] = sum_r xp[r,d]^2 * wrow[r] ----------
__global__ __launch_bounds__(256) void colnorm_kernel(const float* __restrict__ xp,
                                                      const float* __restrict__ wrow,
                                                      float* __restrict__ coln) {
    __shared__ float wl[M_PREFILL];
    for (int i = threadIdx.x; i < M_PREFILL; i += 256) wl[i] = wrow[i];
    __syncthreads();
    const int d = blockIdx.x * 256 + threadIdx.x;
    float s = 0.f;
    for (int r = 0; r < M_PREFILL; ++r) {
        float v = xp[(size_t)r * K_IN + d];
        s += v * v * wl[r];
    }
    coln[d] = s;
}

// ---------- Stage 3: mask[d] = (rank of |coln[d]| < TOPK) -- stable tie-break ----------
__global__ __launch_bounds__(256) void mask_kernel(const float* __restrict__ coln,
                                                   float* __restrict__ maskout) {
    __shared__ float v[K_IN];
    for (int i = threadIdx.x; i < K_IN; i += 256) v[i] = fabsf(coln[i]);
    __syncthreads();
    const int d = blockIdx.x * 256 + threadIdx.x;
    const float vd = v[d];
    int cnt = 0;
    for (int j = 0; j < K_IN; ++j) {
        float vj = v[j];
        cnt += (vj > vd) || (vj == vd && j < d);
    }
    maskout[d] = (cnt < TOPK) ? 1.0f : 0.0f;
}

// ---------- Stage 4: fused [xp ; xd*mask] (2112 x 4096) @ W^T + bias, bf16x3 WMMA ----------
__global__ __launch_bounds__(256)
void griffin_gemm(const float* __restrict__ xp, const float* __restrict__ xd,
                  const float* __restrict__ W,  const float* __restrict__ bias,
                  const float* __restrict__ mask, float* __restrict__ out) {
    __shared__ __bf16 As_hi[BM * LDS_K];
    __shared__ __bf16 As_lo[BM * LDS_K];
    __shared__ __bf16 Bs_hi[BN * LDS_K];
    __shared__ __bf16 Bs_lo[BN * LDS_K];

    const int tid  = threadIdx.x;
    const int lane = tid & 31;
    const int wave = tid >> 5;
    const int n0 = blockIdx.x * BN;
    const int m0 = blockIdx.y * BM;
    const bool decode_blk = (m0 >= M_PREFILL);

    const int wm = wave & 3;   // M offset = wm*32
    const int wn = wave >> 2;  // N offset = wn*64

    v8f acc[2][4];
#pragma unroll
    for (int mt = 0; mt < 2; ++mt)
#pragma unroll
        for (int nt = 0; nt < 4; ++nt) acc[mt][nt] = (v8f){};

    for (int kt = 0; kt < K_IN / BK; ++kt) {
        const int k0 = kt * BK;

        // ---- load A tile (128x32 fp32 -> hi/lo bf16 in LDS) ----
#pragma unroll
        for (int i = 0; i < 4; ++i) {
            const int f  = tid + 256 * i;       // float4 index over 128x8
            const int m  = f >> 3;
            const int kc = (f & 7) << 2;        // column (floats)
            float4 v;
            if (!decode_blk) {
                v = *(const float4*)(xp + (size_t)(m0 + m) * K_IN + k0 + kc);
            } else {
                const int mr = m0 + m - M_PREFILL;
                if (mr < M_DECODE) {
                    v = *(const float4*)(xd + (size_t)mr * K_IN + k0 + kc);
                    const float4 mk = *(const float4*)(mask + k0 + kc);
                    v.x *= mk.x; v.y *= mk.y; v.z *= mk.z; v.w *= mk.w;
                } else {
                    v = make_float4(0.f, 0.f, 0.f, 0.f);
                }
            }
            __bf16* dh = &As_hi[m * LDS_K + kc];
            __bf16* dl = &As_lo[m * LDS_K + kc];
            split_bf16(v.x, dh + 0, dl + 0);
            split_bf16(v.y, dh + 1, dl + 1);
            split_bf16(v.z, dh + 2, dl + 2);
            split_bf16(v.w, dh + 3, dl + 3);
        }

        // ---- load B tile: Bs[n][k] = W[n0+n][k0+k] ----
#pragma unroll
        for (int i = 0; i < 4; ++i) {
            const int f  = tid + 256 * i;
            const int n  = f >> 3;
            const int kc = (f & 7) << 2;
            const float* src = W + (size_t)(n0 + n) * K_IN + k0 + kc;
            float4 v = *(const float4*)src;
            if (kt + 1 < K_IN / BK) __builtin_prefetch(src + BK, 0, 0);  // next K tile -> global_prefetch_b8
            __bf16* dh = &Bs_hi[n * LDS_K + kc];
            __bf16* dl = &Bs_lo[n * LDS_K + kc];
            split_bf16(v.x, dh + 0, dl + 0);
            split_bf16(v.y, dh + 1, dl + 1);
            split_bf16(v.z, dh + 2, dl + 2);
            split_bf16(v.w, dh + 3, dl + 3);
        }
        __syncthreads();

        // ---- fragments + bf16x3 WMMA ----
        v16bf ah[2], al[2], bh[4], bl[4];
#pragma unroll
        for (int mt = 0; mt < 2; ++mt) {
            const int mrow = (wm * 32 + mt * 16) * LDS_K;
            ah[mt] = load_frag_a(&As_hi[mrow], lane);
            al[mt] = load_frag_a(&As_lo[mrow], lane);
        }
#pragma unroll
        for (int nt = 0; nt < 4; ++nt) {
            const int nrow = (wn * 64 + nt * 16) * LDS_K;
            bh[nt] = load_frag_b(&Bs_hi[nrow], lane);
            bl[nt] = load_frag_b(&Bs_lo[nrow], lane);
        }
#pragma unroll
        for (int mt = 0; mt < 2; ++mt) {
#pragma unroll
            for (int nt = 0; nt < 4; ++nt) {
                acc[mt][nt] = __builtin_amdgcn_wmma_f32_16x16x32_bf16(
                    false, ah[mt], false, bh[nt], (short)0, acc[mt][nt], false, false);
                acc[mt][nt] = __builtin_amdgcn_wmma_f32_16x16x32_bf16(
                    false, ah[mt], false, bl[nt], (short)0, acc[mt][nt], false, false);
                acc[mt][nt] = __builtin_amdgcn_wmma_f32_16x16x32_bf16(
                    false, al[mt], false, bh[nt], (short)0, acc[mt][nt], false, false);
            }
        }
        __syncthreads();
    }

    // ---- epilogue: D layout -> lanes 0-15 M=j, lanes 16-31 M=j+8; N = lane%16 ----
    const int half = lane >> 4;
    const int nl   = lane & 15;
#pragma unroll
    for (int mt = 0; mt < 2; ++mt) {
#pragma unroll
        for (int nt = 0; nt < 4; ++nt) {
            const int n = n0 + wn * 64 + nt * 16 + nl;
            const float b = bias[n];
#pragma unroll
            for (int j = 0; j < 8; ++j) {
                const int m = m0 + wm * 32 + mt * 16 + j + 8 * half;
                if (m < M_TOTAL)
                    out[(size_t)m * N_OUT + n] = acc[mt][nt][j] + b;
            }
        }
    }
}

extern "C" void kernel_launch(void* const* d_in, const int* in_sizes, int n_in,
                              void* d_out, int out_size, void* d_ws, size_t ws_size,
                              hipStream_t stream) {
    const float* xp   = (const float*)d_in[0];  // (1,2048,4096)
    const float* xd   = (const float*)d_in[1];  // (64,1,4096)
    const float* W    = (const float*)d_in[2];  // (11008,4096)
    const float* bias = (const float*)d_in[3];  // (11008)
    float* out = (float*)d_out;                 // [out_prefill | out_decode] contiguous
    float* ws  = (float*)d_ws;
    float* wrow = ws;                   // 2048
    float* coln = ws + 2048;            // 4096
    float* mask = ws + 2048 + 4096;     // 4096

    row_w_kernel <<<M_PREFILL, 256, 0, stream>>>(xp, wrow);
    colnorm_kernel<<<K_IN / 256, 256, 0, stream>>>(xp, wrow, coln);
    mask_kernel  <<<K_IN / 256, 256, 0, stream>>>(coln, mask);

    dim3 grid(N_OUT / BN, (M_TOTAL + BM - 1) / BM);  // 86 x 17
    griffin_gemm<<<grid, 256, 0, stream>>>(xp, xd, W, bias, mask, out);
}
